// DKD_27556510171207
// MI455X (gfx1250) — matrix operand order repaired and checked by reading, not compiled, not run on hardware
//
#include <hip/hip_runtime.h>
#include <math.h>

// ---------------------------------------------------------------- constants
#define IMG_H   1024
#define IMG_W   1024
#define IMG_HW  (1024*1024)
#define NIMG    8
#define TOPK    4096
#define CAP     131072            // per-image candidate cap (~3x expected 42K)

// NMS tile geometry: 32x32 output, chain of 5 maxpools(r=2) => halo 10
#define S_LD 52
#define S_M0 48
#define S_S1 44
#define S_M1 40
#define S_S2 36

typedef __attribute__((ext_vector_type(16))) _Float16 v16h;
typedef __attribute__((ext_vector_type(8)))  float    v8f;

#define AS_GLOBAL __attribute__((address_space(1)))
#define AS_LDS    __attribute__((address_space(3)))

#if __has_builtin(__builtin_amdgcn_global_load_async_to_lds_b32) && \
    __has_builtin(__builtin_amdgcn_s_wait_asynccnt)
#define USE_ASYNC_LDS 1
#else
#define USE_ASYNC_LDS 0
#endif

// ---------------------------------------------------------------- kernel 0
__global__ void init_counters(int* cnt) {
    if (threadIdx.x < NIMG) cnt[threadIdx.x] = 0;
}

// ---------------------------------------------------------------- kernel 1
// Fully fused simple_nms (maxpool chain + 2 suppression iters) + border zero.
// One global read + one global write of the score map. All 5x5 maxpools are
// separable (row-max then col-max): ~2.4x fewer LDS reads than direct 5x5.
// Tile fill uses GLOBAL_LOAD_ASYNC_TO_LDS_B32 (ASYNCcnt) when available.
__global__ __launch_bounds__(256) void nms_kernel(const float* __restrict__ in,
                                                  float* __restrict__ outNms) {
    const int b   = blockIdx.z;
    const int ty0 = blockIdx.y * 32;
    const int tx0 = blockIdx.x * 32;
    const float* img = in + (size_t)b * IMG_HW;

    __shared__ float sm  [S_LD*S_LD];     // scores tile, halo 10
    __shared__ float rmax[S_LD*S_M0];     // reusable row-max temp (52x48 max)
    __shared__ float m0a [S_M0*S_M0];
    __shared__ float ss1 [S_S1*S_S1];
    __shared__ float sp1 [S_S1*S_S1];
    __shared__ float m1a [S_M1*S_M1];
    __shared__ float ss2 [S_S2*S_S2];
    __shared__ float sp2 [S_S2*S_S2];

    const int tid = threadIdx.x;

    // ---- load 52x52 tile, zero pad (== -inf pad: all in-bounds values >= 0)
    for (int i = tid; i < S_LD*S_LD; i += 256) {
        int ly = i / S_LD, lx = i - ly * S_LD;
        int gy = ty0 - 10 + ly, gx = tx0 - 10 + lx;
        bool inB = ((unsigned)gy < (unsigned)IMG_H) && ((unsigned)gx < (unsigned)IMG_W);
#if USE_ASYNC_LDS
        if (inB) {
            // direct HBM/L2 -> LDS, no VGPR round trip (tracked by ASYNCcnt)
            __builtin_amdgcn_global_load_async_to_lds_b32(
                (AS_GLOBAL int*)(unsigned long long)(uintptr_t)(img + gy * IMG_W + gx),
                (AS_LDS    int*)(unsigned int)(uintptr_t)&sm[i],
                0, 0);
        } else {
            sm[i] = 0.f;
        }
#else
        float v = 0.f;
        if (inB) v = img[gy * IMG_W + gx];
        sm[i] = v;
#endif
    }
#if USE_ASYNC_LDS
    __builtin_amdgcn_s_wait_asynccnt(0);
#endif
    __syncthreads();

    // ---- row-max of sm (52 rows x 48 cols)
    for (int i = tid; i < S_LD*S_M0; i += 256) {
        int y = i / S_M0, x = i - y * S_M0;
        float mx = sm[y*S_LD + x];
        #pragma unroll
        for (int d = 1; d < 5; ++d) mx = fmaxf(mx, sm[y*S_LD + x + d]);
        rmax[i] = mx;
    }
    __syncthreads();

    // ---- mask0 = (s == maxpool(s))  (halo 8, 48x48)
    for (int i = tid; i < S_M0*S_M0; i += 256) {
        int y = i / S_M0, x = i - y * S_M0;
        float mx = rmax[y*S_M0 + x];
        #pragma unroll
        for (int d = 1; d < 5; ++d) mx = fmaxf(mx, rmax[(y+d)*S_M0 + x]);
        float c = sm[(y+2)*S_LD + (x+2)];
        int gy = ty0 - 8 + y, gx = tx0 - 8 + x;
        bool inB = ((unsigned)gy < (unsigned)IMG_H) && ((unsigned)gx < (unsigned)IMG_W);
        m0a[i] = (inB && c == mx) ? 1.f : 0.f;
    }
    __syncthreads();

    // ---- row-max of mask0 (48 rows x 44 cols)
    for (int i = tid; i < S_M0*S_S1; i += 256) {
        int y = i / S_S1, x = i - y * S_S1;
        float mx = m0a[y*S_M0 + x];
        #pragma unroll
        for (int d = 1; d < 5; ++d) mx = fmaxf(mx, m0a[y*S_M0 + x + d]);
        rmax[y*S_S1 + x] = mx;
    }
    __syncthreads();

    // ---- iter 1: supp1 = maxpool(mask0)>0 ; ss1 = supp1 ? 0 : s  (44x44)
    for (int i = tid; i < S_S1*S_S1; i += 256) {
        int y = i / S_S1, x = i - y * S_S1;
        float mx = rmax[y*S_S1 + x];
        #pragma unroll
        for (int d = 1; d < 5; ++d) mx = fmaxf(mx, rmax[(y+d)*S_S1 + x]);
        bool supp = mx > 0.f;
        float c = sm[(y+4)*S_LD + (x+4)];
        int gy = ty0 - 6 + y, gx = tx0 - 6 + x;
        bool inB = ((unsigned)gy < (unsigned)IMG_H) && ((unsigned)gx < (unsigned)IMG_W);
        sp1[i] = supp ? 1.f : 0.f;
        ss1[i] = (inB && !supp) ? c : 0.f;
    }
    __syncthreads();

    // ---- row-max of ss1 (44 rows x 40 cols)
    for (int i = tid; i < S_S1*S_M1; i += 256) {
        int y = i / S_M1, x = i - y * S_M1;
        float mx = ss1[y*S_S1 + x];
        #pragma unroll
        for (int d = 1; d < 5; ++d) mx = fmaxf(mx, ss1[y*S_S1 + x + d]);
        rmax[y*S_M1 + x] = mx;
    }
    __syncthreads();

    // ---- mask1 = mask0 | (ss1==maxpool(ss1) & ~supp1)  (40x40)
    for (int i = tid; i < S_M1*S_M1; i += 256) {
        int y = i / S_M1, x = i - y * S_M1;
        float mx = rmax[y*S_M1 + x];
        #pragma unroll
        for (int d = 1; d < 5; ++d) mx = fmaxf(mx, rmax[(y+d)*S_M1 + x]);
        float c   = ss1[(y+2)*S_S1 + (x+2)];
        bool newm = (c == mx);
        bool sup  = sp1[(y+2)*S_S1 + (x+2)] > 0.f;
        bool prev = m0a[(y+4)*S_M0 + (x+4)] > 0.f;
        bool mres = prev || (newm && !sup);
        int gy = ty0 - 4 + y, gx = tx0 - 4 + x;
        bool inB = ((unsigned)gy < (unsigned)IMG_H) && ((unsigned)gx < (unsigned)IMG_W);
        m1a[i] = (inB && mres) ? 1.f : 0.f;
    }
    __syncthreads();

    // ---- row-max of mask1 (40 rows x 36 cols)
    for (int i = tid; i < S_M1*S_S2; i += 256) {
        int y = i / S_S2, x = i - y * S_S2;
        float mx = m1a[y*S_M1 + x];
        #pragma unroll
        for (int d = 1; d < 5; ++d) mx = fmaxf(mx, m1a[y*S_M1 + x + d]);
        rmax[y*S_S2 + x] = mx;
    }
    __syncthreads();

    // ---- iter 2: supp2 / ss2  (36x36)
    for (int i = tid; i < S_S2*S_S2; i += 256) {
        int y = i / S_S2, x = i - y * S_S2;
        float mx = rmax[y*S_S2 + x];
        #pragma unroll
        for (int d = 1; d < 5; ++d) mx = fmaxf(mx, rmax[(y+d)*S_S2 + x]);
        bool supp = mx > 0.f;
        float c = sm[(y+8)*S_LD + (x+8)];
        int gy = ty0 - 2 + y, gx = tx0 - 2 + x;
        bool inB = ((unsigned)gy < (unsigned)IMG_H) && ((unsigned)gx < (unsigned)IMG_W);
        sp2[i] = supp ? 1.f : 0.f;
        ss2[i] = (inB && !supp) ? c : 0.f;
    }
    __syncthreads();

    // ---- row-max of ss2 (36 rows x 32 cols)
    for (int i = tid; i < S_S2*32; i += 256) {
        int y = i / 32, x = i - y * 32;
        float mx = ss2[y*S_S2 + x];
        #pragma unroll
        for (int d = 1; d < 5; ++d) mx = fmaxf(mx, ss2[y*S_S2 + x + d]);
        rmax[y*32 + x] = mx;
    }
    __syncthreads();

    // ---- final mask2, nms output, border zero (32x32)
    for (int i = tid; i < 32*32; i += 256) {
        int y = i / 32, x = i - y * 32;
        float mx = rmax[y*32 + x];
        #pragma unroll
        for (int d = 1; d < 5; ++d) mx = fmaxf(mx, rmax[(y+d)*32 + x]);
        float c   = ss2[(y+2)*S_S2 + (x+2)];
        bool newm = (c == mx);
        bool sup  = sp2[(y+2)*S_S2 + (x+2)] > 0.f;
        bool prev = m1a[(y+4)*S_M1 + (x+4)] > 0.f;
        bool mres = prev || (newm && !sup);
        int gy = ty0 + y, gx = tx0 + x;
        float sv = sm[(y+10)*S_LD + (x+10)];
        bool keep = mres && gy >= 2 && gy < (IMG_H-2) && gx >= 2 && gx < (IMG_W-2);
        outNms[(size_t)b * IMG_HW + gy * IMG_W + gx] = keep ? sv : 0.f;
    }
}

// ---------------------------------------------------------------- kernel 2
// Compact nonzero survivors (density ~1/25) into per-image candidate lists.
__global__ __launch_bounds__(256) void compact_kernel(const float* __restrict__ nmsv,
                                                      int* __restrict__ cnt,
                                                      float* __restrict__ cv,
                                                      int* __restrict__ ci) {
    int gid = blockIdx.x * 256 + threadIdx.x;          // < 8 * 2^20
    __builtin_prefetch(nmsv + gid + 8192, 0, 0);       // global_prefetch_b8
    float v = nmsv[gid];
    if (v > 0.f) {
        int b = gid >> 20;
        int p = gid & (IMG_HW - 1);
        int pos = atomicAdd(&cnt[b], 1);
        if (pos < CAP) {
            cv[(size_t)b * CAP + pos] = v;
            ci[(size_t)b * CAP + pos] = p;
        }
    }
}

// ---------------------------------------------------------------- kernel 3
// Per image: 4-level radix select (fp32 bits, all values >= 0) for the exact
// top-4096 threshold, compact into LDS, bitonic-sort descending with key
// (valBits<<32)|~idx to reproduce lax.top_k ordering (ties -> lower index).
__global__ __launch_bounds__(1024) void select_sort_kernel(const float* __restrict__ cv,
                                                           const int* __restrict__ ci,
                                                           const int* __restrict__ cnt,
                                                           int* __restrict__ topIdx) {
    const int b   = blockIdx.x;
    const int tid = threadIdx.x;
    __shared__ unsigned int hist[256];
    __shared__ unsigned long long pairs[TOPK];
    __shared__ unsigned int sh_prefix, sh_krem, sh_above, sh_tie;

    int n = cnt[b];
    if (n > CAP) n = CAP;
    const float* val = cv + (size_t)b * CAP;
    const int*   idx = ci + (size_t)b * CAP;

    // prefill: value 0, index 0 (key low word = ~0 -> idx 0)
    for (int i = tid; i < TOPK; i += 1024) pairs[i] = 0xFFFFFFFFull;
    if (tid == 0) { sh_prefix = 0u; sh_krem = TOPK; }
    __syncthreads();

    if (n > TOPK) {
        for (int shift = 24; shift >= 0; shift -= 8) {
            for (int i = tid; i < 256; i += 1024) hist[i] = 0u;
            __syncthreads();
            unsigned int pre   = sh_prefix;
            unsigned int upper = (shift == 24) ? 0u : (0xFFFFFFFFu << (shift + 8));
            for (int i = tid; i < n; i += 1024) {
                unsigned int bits = __float_as_uint(val[i]);
                if ((bits & upper) == pre)
                    atomicAdd(&hist[(bits >> shift) & 255u], 1u);
            }
            __syncthreads();
            if (tid == 0) {
                unsigned int k = sh_krem, acc = 0u;
                int bin = 255;
                for (; bin > 0; --bin) {
                    if (acc + hist[bin] >= k) break;
                    acc += hist[bin];
                }
                sh_prefix = pre | ((unsigned int)bin << shift);
                sh_krem   = k - acc;
            }
            __syncthreads();
        }
    }

    unsigned int T      = (n > TOPK) ? sh_prefix : 0u;
    unsigned int kremF  = (n > TOPK) ? sh_krem   : 0u;   // ties taken at T
    unsigned int tieStart = TOPK - kremF;                // = count strictly > T
    if (tid == 0) { sh_above = 0u; sh_tie = 0u; }
    __syncthreads();

    for (int i = tid; i < n; i += 1024) {
        unsigned int bits = __float_as_uint(val[i]);
        unsigned long long key =
            ((unsigned long long)bits << 32) | (unsigned int)(~idx[i]);
        if (bits > T) {
            unsigned int s0 = atomicAdd(&sh_above, 1u);
            if (s0 < tieStart) pairs[s0] = key;
        } else if (bits == T && T != 0u) {
            unsigned int t0 = atomicAdd(&sh_tie, 1u);
            if (t0 < kremF) pairs[tieStart + t0] = key;
        }
    }
    __syncthreads();

    // bitonic sort, descending
    for (unsigned int k = 2; k <= TOPK; k <<= 1) {
        for (unsigned int j = k >> 1; j > 0; j >>= 1) {
            for (unsigned int i = tid; i < TOPK; i += 1024) {
                unsigned int ixj = i ^ j;
                if (ixj > i) {
                    unsigned long long A = pairs[i], Bv = pairs[ixj];
                    bool descBlock = ((i & k) == 0u);
                    if (descBlock ? (A < Bv) : (A > Bv)) {
                        pairs[i] = Bv; pairs[ixj] = A;
                    }
                }
            }
            __syncthreads();
        }
    }

    for (int i = tid; i < TOPK; i += 1024)
        topIdx[b * TOPK + i] = (int)(~((unsigned int)(pairs[i] & 0xFFFFFFFFull)));
}

// ---------------------------------------------------------------- kernel 4
// Patch gather + soft-argmax. Each wave32 handles 16 keypoints: lane L owns
// taps K in {0..7,16..23}, lane L+16 owns {8..15,24} (ISA 16-bit A layout).
// One V_WMMA_F32_16X16X32_F16 computes [Sx, Sy, S0, Sq] for 16 keypoints:
//   A = x_exp (16x25 padded to 16x32 f16), B cols = [dx, dy, 1, dx^2+dy^2].
__global__ __launch_bounds__(256) void patch_kernel(const float* __restrict__ scores,
                                                    const int* __restrict__ topIdx,
                                                    float* __restrict__ out) {
    const int tid  = threadIdx.x;
    const int lane = tid & 31;
    const int wv   = tid >> 5;                 // 0..7 waves/block
    const int m    = lane & 15;                // keypoint row within tile
    const int half = lane >> 4;                // K-half selector
    const int kp   = (blockIdx.x * 8 + wv) * 16 + m;   // 0..32767
    const int b    = kp >> 12;
    const int idx  = topIdx[kp];
    const int ky   = idx >> 10;
    const int kx   = idx & 1023;
    const float* img = scores + (size_t)b * IMG_HW;

    float vals[16];
    float localMax = -1e30f;
    #pragma unroll
    for (int j = 0; j < 16; ++j) {
        int K = (j < 8) ? (half * 8 + j) : (16 + half * 8 + (j - 8));
        float v = 0.f;
        if (K < 25) {
            int dy = K / 5 - 2, dx = K % 5 - 2;
            int y = ky + dy, x = kx + dx;
            if ((unsigned)y < (unsigned)IMG_H && (unsigned)x < (unsigned)IMG_W)
                v = img[y * IMG_W + x];
            localMax = fmaxf(localMax, v);     // zero-padded taps join the max
        }
        vals[j] = v;
    }
    float om = __shfl_xor(localMax, 16, 32);
    float mx = fmaxf(localMax, om);

    v16h a, bm;
    #pragma unroll
    for (int j = 0; j < 16; ++j) {
        int K = (j < 8) ? (half * 8 + j) : (16 + half * 8 + (j - 8));
        float e = (K < 25) ? __expf((vals[j] - mx) * 10.f) : 0.f;   // /TEMP=0.1
        a[j] = (_Float16)e;
        int Kb = (j < 8) ? (half * 8 + j) : (16 + half * 8 + (j - 8));                 // B layout: lane n = column N
        float bv = 0.f;
        if (Kb < 25) {
            float dx = (float)(Kb % 5) - 2.f;
            float dy = (float)(Kb / 5) - 2.f;
            bv = (m == 0) ? dx : (m == 1) ? dy : (m == 2) ? 1.f
               : (m == 3) ? (dx * dx + dy * dy) : 0.f;
        }
        bm[j] = (_Float16)bv;
    }

    v8f c = {};
    c = __builtin_amdgcn_wmma_f32_16x16x32_f16(false, a, false, bm,
                                               (short)0, c, false, false);

    __shared__ float stage[8][32][8];
    #pragma unroll
    for (int i = 0; i < 8; ++i) stage[wv][lane][i] = c[i];
    __syncthreads();

    if (half == 0) {
        // D[M][N]: M<8 -> (vgpr M, lane N); M>=8 -> (vgpr M-8, lane 16+N)
        int srcBase = (m < 8) ? 0 : 16;
        int vg = m & 7;
        float Sx = stage[wv][srcBase + 0][vg];
        float Sy = stage[wv][srcBase + 1][vg];
        float S0 = stage[wv][srcBase + 2][vg];   // >= 1 (max tap gives e^0)
        float Sq = stage[wv][srcBase + 3][vg];
        float inv = 1.f / S0;                    // single divide, reused 3x
        float rx = Sx * inv, ry = Sy * inv;
        float disp = (Sq * inv - (rx * rx + ry * ry)) * 0.25f;
        float xn = ((float)kx + rx) / 1023.f * 2.f - 1.f;
        float yn = ((float)ky + ry) / 1023.f * 2.f - 1.f;

        // bilinear grid_sample, align_corners=True (round-trip like reference)
        float px = (xn + 1.f) * 0.5f * 1023.f;
        float py = (yn + 1.f) * 0.5f * 1023.f;
        float fx = floorf(px), fy = floorf(py);
        float wx = px - fx,   wy = py - fy;
        int x0 = min(max((int)fx, 0), IMG_W - 1);
        int x1 = min(max((int)fx + 1, 0), IMG_W - 1);
        int y0 = min(max((int)fy, 0), IMG_H - 1);
        int y1 = min(max((int)fy + 1, 0), IMG_H - 1);
        float v00 = img[y0 * IMG_W + x0], v01 = img[y0 * IMG_W + x1];
        float v10 = img[y1 * IMG_W + x0], v11 = img[y1 * IMG_W + x1];
        float sc = v00 * (1.f - wx) * (1.f - wy) + v01 * wx * (1.f - wy)
                 + v10 * (1.f - wx) * wy        + v11 * wx * wy;

        out[kp * 2 + 0] = xn;                       // keypoints [8,4096,2]
        out[kp * 2 + 1] = yn;
        out[NIMG * TOPK * 2 + kp] = disp;           // scoredispersity
        out[NIMG * TOPK * 2 + NIMG * TOPK + kp] = sc;   // kptscores
    }
}

// ---------------------------------------------------------------- launcher
extern "C" void kernel_launch(void* const* d_in, const int* in_sizes, int n_in,
                              void* d_out, int out_size, void* d_ws, size_t ws_size,
                              hipStream_t stream) {
    (void)in_sizes; (void)n_in; (void)out_size; (void)ws_size;
    const float* scores = (const float*)d_in[0];
    float* out = (float*)d_out;

    char* ws = (char*)d_ws;
    const size_t NMS_BYTES  = (size_t)NIMG * IMG_HW * sizeof(float);   // 32 MB
    const size_t CNT_BYTES  = 256;
    const size_t CAND_BYTES = (size_t)NIMG * CAP * sizeof(float);      // 4 MB
    float* nmsBuf = (float*)ws;
    int*   cnt    = (int*)  (ws + NMS_BYTES);
    float* cv     = (float*)(ws + NMS_BYTES + CNT_BYTES);
    int*   ci     = (int*)  (ws + NMS_BYTES + CNT_BYTES + CAND_BYTES);
    int*   topIdx = (int*)  (ws + NMS_BYTES + CNT_BYTES + 2 * CAND_BYTES);

    init_counters<<<1, 32, 0, stream>>>(cnt);

    dim3 gNms(IMG_W / 32, IMG_H / 32, NIMG);
    nms_kernel<<<gNms, 256, 0, stream>>>(scores, nmsBuf);

    compact_kernel<<<(NIMG * IMG_HW) / 256, 256, 0, stream>>>(nmsBuf, cnt, cv, ci);

    select_sort_kernel<<<NIMG, 1024, 0, stream>>>(cv, ci, cnt, topIdx);

    patch_kernel<<<(NIMG * TOPK) / 128, 256, 0, stream>>>(scores, topIdx, out);
}
